// PointNetAbstract_53712861004286
// MI455X (gfx1250) — matrix operand order repaired and checked by reading, not compiled, not run on hardware
//
#include <hip/hip_runtime.h>
#include <hip/hip_bf16.h>
#include <stdint.h>

typedef __attribute__((ext_vector_type(16))) _Float16 v16h;
typedef __attribute__((ext_vector_type(8)))  _Float16 v8h;
typedef __attribute__((ext_vector_type(8)))  float    v8f;

// ---------------------------------------------------------------------------
// CDNA5 async global->LDS helpers (ASYNCcnt path; no builtin coverage -> asm)
// ---------------------------------------------------------------------------
__device__ __forceinline__ void async_load_b128_to_lds(uint32_t lds_off, const void* gptr) {
  asm volatile("global_load_async_to_lds_b128 %0, %1, off"
               :: "v"(lds_off), "v"((unsigned long long)(uintptr_t)gptr)
               : "memory");
}
__device__ __forceinline__ void wait_async0() {
  asm volatile("s_wait_asynccnt 0x0" ::: "memory");
}

// ---------------------------------------------------------------------------
// Kernels
// ---------------------------------------------------------------------------

// xyz (8,3,4096) -> pts (8,4096,3)
__global__ void k_transpose(const float* __restrict__ xin, float* __restrict__ pts) {
  int i = blockIdx.x * blockDim.x + threadIdx.x;
  if (i >= 8 * 4096) return;
  int b = i >> 12, n = i & 4095;
  const float* s = xin + (size_t)b * 3 * 4096 + n;
  float* d = pts + (size_t)i * 3;
  d[0] = s[0]; d[1] = s[4096]; d[2] = s[8192];
}

// Farthest point sampling: one block per batch. dist array lives in LDS.
__global__ __launch_bounds__(256) void k_fps(const float* __restrict__ xyz, int N, int S,
                                             int* __restrict__ fps_idx, float* __restrict__ nxyz) {
  __shared__ float dist[4096];
  __shared__ float rv[256];
  __shared__ int   ri[256];
  __shared__ int   sfar;
  const int b = blockIdx.x, t = threadIdx.x;
  const float* X = xyz + (size_t)b * N * 3;
  for (int i = t; i < N; i += 256) dist[i] = 1e10f;
  __syncthreads();
  int far = 0;
  for (int it = 0; it < S; ++it) {
    float cx = X[far * 3], cy = X[far * 3 + 1], cz = X[far * 3 + 2];
    if (t == 0) {
      fps_idx[b * S + it] = far;
      float* q = nxyz + ((size_t)b * S + it) * 3;
      q[0] = cx; q[1] = cy; q[2] = cz;
    }
    float best = -1.0f; int bi = 0;
    for (int i = t; i < N; i += 256) {
      float dx = X[i * 3] - cx, dy = X[i * 3 + 1] - cy, dz = X[i * 3 + 2] - cz;
      float d = dx * dx + dy * dy + dz * dz;
      float old = dist[i];
      d = fminf(old, d);
      dist[i] = d;
      if (d > best) { best = d; bi = i; }
    }
    rv[t] = best; ri[t] = bi;
    __syncthreads();
    for (int s = 128; s > 0; s >>= 1) {
      if (t < s) {
        if (rv[t + s] > rv[t] || (rv[t + s] == rv[t] && ri[t + s] < ri[t])) {
          rv[t] = rv[t + s]; ri[t] = ri[t + s];
        }
      }
      __syncthreads();
    }
    if (t == 0) sfar = ri[0];
    __syncthreads();
    far = sfar;
  }
}

// Ball query: first 32 indices (ascending) with d^2 <= r^2, padded with first hit.
__global__ void k_ball(const float* __restrict__ xyz, const float* __restrict__ nxyz,
                       int N, int S, float r2, int* __restrict__ idx, size_t total) {
  size_t i = (size_t)blockIdx.x * blockDim.x + threadIdx.x;
  if (i >= total) return;
  int b = (int)(i / S);
  const float* X = xyz + (size_t)b * N * 3;
  float qx = nxyz[i * 3], qy = nxyz[i * 3 + 1], qz = nxyz[i * 3 + 2];
  int* o = idx + i * 32;
  int cnt = 0;
  for (int j = 0; j < N; ++j) {
    float dx = X[j * 3] - qx, dy = X[j * 3 + 1] - qy, dz = X[j * 3 + 2] - qz;
    if (dx * dx + dy * dy + dz * dz <= r2) {
      o[cnt++] = j;
      if (cnt == 32) break;
    }
  }
  if (cnt == 0) { o[0] = 0; cnt = 1; }
  int f = o[0];
  for (int c = cnt; c < 32; ++c) o[c] = f;
}

// Build grouped feature rows: A[row, :] = [g_xyz(3) | g_pts(Cprev) | 0-pad], fp16.
__global__ void k_feat(const float* __restrict__ xyz, const float* __restrict__ nxyz,
                       const float* __restrict__ pprev, const int* __restrict__ idx,
                       int N, int S, int Cprev, int Kpad, _Float16* __restrict__ A, size_t total) {
  size_t row = (size_t)blockIdx.x * blockDim.x + threadIdx.x;
  if (row >= total) return;
  size_t bs = row >> 5;
  int b = (int)(bs / S);
  int j = idx[row];
  const float* p = xyz + ((size_t)b * N + j) * 3;
  const float* q = nxyz + bs * 3;
  const float* f = pprev + ((size_t)b * N + j) * Cprev;
  _Float16* a = A + row * (size_t)Kpad;
  a[0] = (_Float16)(p[0] - q[0]);
  a[1] = (_Float16)(p[1] - q[1]);
  a[2] = (_Float16)(p[2] - q[2]);
  int c = 0;
  for (; c < Cprev; ++c) a[3 + c] = (_Float16)f[c];
  for (c += 3; c < Kpad; ++c) a[c] = (_Float16)0.0f;
}

// Pack weights W (Cout x Korig, f32 row-major) into WMMA B-fragment order, fp16.
// Bp[((nt*Ktiles + kt)*32 + lane)*16 + j] = W[nt*16 + (lane&15)][kt*32 + 8*(lane>>4) + (j&7) + 16*(j>>3)]
__global__ void k_pack_w(const float* __restrict__ W, _Float16* __restrict__ Bp,
                         int Korig, int Norig, int Ktiles, int Ntiles, size_t total) {
  size_t i = (size_t)blockIdx.x * blockDim.x + threadIdx.x;
  if (i >= total) return;
  int j = (int)(i & 15);
  int l = (int)((i >> 4) & 31);
  size_t t2 = i >> 9;
  int kt = (int)(t2 % Ktiles);
  int nt = (int)(t2 / Ktiles);
  int k = kt * 32 + ((l >> 4) << 3) + (j & 7) + ((j >> 3) << 4);
  int n = nt * 16 + (l & 15);
  float v = (k < Korig && n < Norig) ? W[(size_t)n * Korig + k] : 0.0f;
  Bp[i] = (_Float16)v;
}

// Tiled GEMM: Out(MxN) = A(MxKpad,f16) * Bp(packed,f16) + bias, f32 accumulate.
// 8 waves per block along M; each wave computes NT 16x16 output tiles sharing
// one A fragment. B fragments are staged block-wide into LDS with CDNA5
// async global->LDS copies (double buffered, ASYNCcnt).
template <int NT>
__global__ __launch_bounds__(256) void k_gemm_wmma(
    const _Float16* __restrict__ A, const _Float16* __restrict__ Bp,
    const float* __restrict__ bias, float* __restrict__ Out,
    int Mtiles, int Ktiles, int Kpad, int Nstride) {
  __shared__ _Float16 lbuf[2][NT * 512];

  const int tid  = threadIdx.x;
  const int wid  = tid >> 5;
  const int lane = tid & 31;
  const int mtile = blockIdx.x * 8 + wid;
  const int nbase = blockIdx.y * NT;
  const bool active = (mtile < Mtiles);   // wave-uniform
  const int g = lane >> 4;
  const int h = lane & 15;

  // Staging role: thread t stages 16B unit (t&63) of ntile chunk (t>>6).
  const int snt   = tid >> 6;
  const int selem = tid & 63;
  const bool stager = (tid < NT * 64);
  const _Float16* gsrc = Bp + ((size_t)(nbase + snt) * Ktiles) * 512 + (size_t)selem * 8;
  const uint32_t lds0 = (uint32_t)(uintptr_t)&lbuf[0][snt * 512 + selem * 8];
  const uint32_t lds1 = (uint32_t)(uintptr_t)&lbuf[1][snt * 512 + selem * 8];

  const _Float16* Ar = A + (size_t)(mtile * 16 + h) * Kpad + 8 * g;

  v8f acc[NT];
#pragma unroll
  for (int t = 0; t < NT; ++t) acc[t] = (v8f){};

  // Prologue: stage kt=0 into buffer 0.
  if (stager) async_load_b128_to_lds(lds0, gsrc);
  wait_async0();
  __syncthreads();

  for (int kt = 0; kt < Ktiles; ++kt) {
    const int cur = kt & 1;
    if (stager && (kt + 1 < Ktiles)) {
      async_load_b128_to_lds(cur ? lds0 : lds1, gsrc + (size_t)(kt + 1) * 512);
    }
    if (active) {
      v8h a0 = *(const v8h*)(Ar + (size_t)kt * 32);
      v8h a1 = *(const v8h*)(Ar + (size_t)kt * 32 + 16);
      v16h a;
#pragma unroll
      for (int j = 0; j < 8; ++j) { a[j] = a0[j]; a[j + 8] = a1[j]; }
      if (kt + 2 < Ktiles) __builtin_prefetch(Ar + (size_t)(kt + 2) * 32, 0, 3);
#pragma unroll
      for (int t = 0; t < NT; ++t) {
        v16h b = *(const v16h*)&lbuf[cur][t * 512 + lane * 16];
        acc[t] = __builtin_amdgcn_wmma_f32_16x16x32_f16(
            /*neg_a=*/false, a, /*neg_b=*/false, b,
            /*c_mod=*/(short)0, acc[t], /*reuse_a=*/false, /*reuse_b=*/false);
      }
    }
    wait_async0();
    __syncthreads();
  }

  if (active) {
    float* Orow = Out + (size_t)(mtile * 16 + 8 * g) * Nstride;
#pragma unroll
    for (int t = 0; t < NT; ++t) {
      const int n = (nbase + t) * 16 + h;
      const float bv = bias[n];
#pragma unroll
      for (int r = 0; r < 8; ++r) Orow[(size_t)r * Nstride + n] = acc[t][r] + bv;
    }
  }
}

// Per-channel mean/var over M rows (population variance). One block per channel.
__global__ __launch_bounds__(256) void k_bnstats(const float* __restrict__ Out, size_t M,
                                                 int C, float* __restrict__ stats) {
  int c = blockIdx.x, t = threadIdx.x;
  __shared__ float s1[256], s2[256];
  float a = 0.0f, q = 0.0f;
  for (size_t m = t; m < M; m += 256) {
    float x = Out[m * C + c];
    a += x; q += x * x;
  }
  s1[t] = a; s2[t] = q;
  __syncthreads();
  for (int s = 128; s > 0; s >>= 1) {
    if (t < s) { s1[t] += s1[t + s]; s2[t] += s2[t + s]; }
    __syncthreads();
  }
  if (t == 0) {
    float mu = s1[0] / (float)M;
    stats[2 * c] = mu;
    stats[2 * c + 1] = s2[0] / (float)M - mu * mu;
  }
}

// BN + ReLU, write fp16 rows (padded) for the next GEMM.
__global__ void k_bnapply(const float* __restrict__ Out, const float* __restrict__ stats,
                          const float* __restrict__ gamma, const float* __restrict__ beta,
                          int C, int Nstride, int KpadNext, _Float16* __restrict__ An, size_t total) {
  size_t i = (size_t)blockIdx.x * blockDim.x + threadIdx.x;
  if (i >= total) return;
  size_t m = i / KpadNext;
  int c = (int)(i % KpadNext);
  float v = 0.0f;
  if (c < C) {
    float x = Out[m * Nstride + c];
    x = (x - stats[2 * c]) * rsqrtf(stats[2 * c + 1] + 1e-5f) * gamma[c] + beta[c];
    v = fmaxf(x, 0.0f);
  }
  An[i] = (_Float16)v;
}

// Fused BN + ReLU + max over the 32 group samples -> points (B,S,C) f32.
__global__ void k_maxpool_bn(const float* __restrict__ Out, const float* __restrict__ stats,
                             const float* __restrict__ gamma, const float* __restrict__ beta,
                             int C, float* __restrict__ pts, size_t total) {
  size_t i = (size_t)blockIdx.x * blockDim.x + threadIdx.x;
  if (i >= total) return;
  size_t bs = i / C;
  int c = (int)(i % C);
  float mu = stats[2 * c];
  float inv = rsqrtf(stats[2 * c + 1] + 1e-5f) * gamma[c];
  float bb = beta[c];
  const float* p = Out + (bs * 32) * (size_t)C + c;
  float m = 0.0f;  // ReLU'd values are >= 0
  for (int k = 0; k < 32; ++k) {
    float x = (p[(size_t)k * C] - mu) * inv + bb;
    m = fmaxf(m, fmaxf(x, 0.0f));
  }
  pts[i] = m;
}

// flat[b][c*16+s] = l4p[b][s][c]; rows 8..15 zero (M padded to one WMMA tile).
__global__ void k_flat(const float* __restrict__ l4p, _Float16* __restrict__ A, size_t total) {
  size_t i = (size_t)blockIdx.x * blockDim.x + threadIdx.x;
  if (i >= total) return;
  int row = (int)(i >> 13);
  int col = (int)(i & 8191);
  float v = 0.0f;
  if (row < 8) {
    int c = col >> 4, s = col & 15;
    v = l4p[((size_t)row * 16 + s) * 512 + c];
  }
  A[i] = (_Float16)v;
}

__global__ void k_cvt(const float* __restrict__ Out, _Float16* __restrict__ A, size_t total) {
  size_t i = (size_t)blockIdx.x * blockDim.x + threadIdx.x;
  if (i >= total) return;
  A[i] = (_Float16)Out[i];
}

// Final BN over the batch dim (8 rows) + ReLU -> d_out (8,512).
__global__ void k_finalbn(const float* __restrict__ h, const float* __restrict__ g,
                          const float* __restrict__ be, float* __restrict__ out) {
  int c = blockIdx.x * blockDim.x + threadIdx.x;
  if (c >= 512) return;
  float s = 0.0f, q = 0.0f;
  for (int b = 0; b < 8; ++b) {
    float x = h[b * 512 + c];
    s += x; q += x * x;
  }
  float mu = s * 0.125f;
  float va = q * 0.125f - mu * mu;
  float inv = rsqrtf(va + 1e-5f) * g[c];
  for (int b = 0; b < 8; ++b) {
    float x = (h[b * 512 + c] - mu) * inv + be[c];
    out[b * 512 + c] = fmaxf(x, 0.0f);
  }
}

// ---------------------------------------------------------------------------
// Host orchestration
// ---------------------------------------------------------------------------

static inline unsigned nblk(size_t total) { return (unsigned)((total + 255) / 256); }

extern "C" void kernel_launch(void* const* d_in, const int* in_sizes, int n_in,
                              void* d_out, int out_size, void* d_ws, size_t ws_size,
                              hipStream_t stream) {
  (void)in_sizes; (void)n_in; (void)out_size; (void)ws_size;

  // Workspace bump allocator (256B aligned).
  char* ws = (char*)d_ws;
  size_t off = 0;
  auto alloc = [&](size_t bytes) -> char* {
    char* p = ws + off;
    off = (off + bytes + 255) & ~(size_t)255;
    return p;
  };

  float*    pts0    = (float*)alloc((size_t)8 * 4096 * 3 * 4);
  float*    nxyzA   = (float*)alloc((size_t)8 * 1024 * 3 * 4);
  float*    nxyzB   = (float*)alloc((size_t)8 * 1024 * 3 * 4);
  float*    poutA   = (float*)alloc((size_t)8 * 1024 * 64 * 4);
  float*    poutB   = (float*)alloc((size_t)8 * 1024 * 64 * 4);
  int*      fpsidx  = (int*)alloc((size_t)8 * 1024 * 4);
  int*      ballidx = (int*)alloc((size_t)8 * 1024 * 32 * 4);
  _Float16* Abuf0   = (_Float16*)alloc((size_t)8 * 1024 * 32 * 96 * 2);   // max M*Kpad halves
  _Float16* Abuf1   = (_Float16*)alloc((size_t)8 * 1024 * 32 * 96 * 2);
  float*    Outbuf  = (float*)alloc((size_t)8 * 1024 * 32 * 64 * 4);      // max M*Cout f32
  _Float16* Wpack   = (_Float16*)alloc((size_t)16777216 * 2);             // FC1: 256*128*512 halves
  float*    stats   = (float*)alloc((size_t)1024 * 4);
  _Float16* Abufs[2] = { Abuf0, Abuf1 };

  const float* xin = (const float*)d_in[0];
  k_transpose<<<nblk((size_t)8 * 4096), 256, 0, stream>>>(xin, pts0);

  const int   Ns[4] = {4096, 1024, 256, 64};
  const int   Ss[4] = {1024, 256, 64, 16};
  const int   Cp[4] = {3, 64, 128, 256};
  const float R2[4] = {0.01f, 0.04f, 0.16f, 0.64f};
  const int   CO[4][3] = {{32, 32, 64}, {64, 64, 128}, {128, 128, 256}, {256, 256, 512}};
  const int   baseIdx[4] = {1, 13, 25, 37};  // sa1..sa4: (W,b,g,be) x3 each

  auto gemm = [&](const _Float16* Ain, const float* W, const float* bias,
                  size_t M, int Korig, int Kpad, int Cout) {
    int Ktiles = Kpad / 32, Ntiles = Cout / 16;
    size_t wtot = (size_t)Ktiles * Ntiles * 512;
    k_pack_w<<<nblk(wtot), 256, 0, stream>>>(W, Wpack, Korig, Cout, Ktiles, Ntiles, wtot);
    int Mtiles = (int)(M / 16);
    unsigned gx = (unsigned)((Mtiles + 7) / 8);
    if ((Ntiles & 3) == 0) {
      dim3 gg(gx, (unsigned)(Ntiles / 4));
      k_gemm_wmma<4><<<gg, 256, 0, stream>>>(Ain, Wpack, bias, Outbuf, Mtiles, Ktiles, Kpad, Cout);
    } else {
      dim3 gg(gx, (unsigned)(Ntiles / 2));
      k_gemm_wmma<2><<<gg, 256, 0, stream>>>(Ain, Wpack, bias, Outbuf, Mtiles, Ktiles, Kpad, Cout);
    }
  };

  const float* xyz_cur = pts0;
  const float* points_cur = pts0;

  for (int l = 0; l < 4; ++l) {
    const int S = Ss[l], N = Ns[l], Cprev = Cp[l];
    float* nx = (l & 1) ? nxyzB : nxyzA;
    float* po = (l & 1) ? poutB : poutA;

    k_fps<<<8, 256, 0, stream>>>(xyz_cur, N, S, fpsidx, nx);
    {
      size_t tot = (size_t)8 * S;
      k_ball<<<nblk(tot), 256, 0, stream>>>(xyz_cur, nx, N, S, R2[l], ballidx, tot);
    }
    const int Cin = 3 + Cprev;
    const int Kpad0 = (Cin + 31) & ~31;
    const size_t M = (size_t)8 * S * 32;
    k_feat<<<nblk(M), 256, 0, stream>>>(xyz_cur, nx, points_cur, ballidx, N, S, Cprev, Kpad0,
                                        Abufs[0], M);
    const _Float16* Ain = Abufs[0];
    int Kpad = Kpad0, Korig = Cin;
    for (int j = 0; j < 3; ++j) {
      const float* W  = (const float*)d_in[baseIdx[l] + 4 * j + 0];
      const float* bb = (const float*)d_in[baseIdx[l] + 4 * j + 1];
      const float* gg = (const float*)d_in[baseIdx[l] + 4 * j + 2];
      const float* be = (const float*)d_in[baseIdx[l] + 4 * j + 3];
      const int Cout = CO[l][j];
      gemm(Ain, W, bb, M, Korig, Kpad, Cout);
      k_bnstats<<<(unsigned)Cout, 256, 0, stream>>>(Outbuf, M, Cout, stats);
      if (j < 2) {
        _Float16* An = (Ain == Abufs[0]) ? Abufs[1] : Abufs[0];
        size_t tot = M * (size_t)Cout;  // Cout is a multiple of 32 -> KpadNext == Cout
        k_bnapply<<<nblk(tot), 256, 0, stream>>>(Outbuf, stats, gg, be, Cout, Cout, Cout, An, tot);
        Ain = An; Korig = Cout; Kpad = Cout;
      } else {
        size_t tot = (size_t)8 * S * Cout;
        k_maxpool_bn<<<nblk(tot), 256, 0, stream>>>(Outbuf, stats, gg, be, Cout, po, tot);
        points_cur = po;
      }
    }
    xyz_cur = nx;
  }

  // FC head: flat (16x8192, rows 8..15 zero) -> 2048 -> 1024 -> 512 -> BN(batch) -> ReLU
  {
    size_t tot = (size_t)16 * 8192;
    k_flat<<<nblk(tot), 256, 0, stream>>>(points_cur, Abufs[0], tot);
  }
  const float* W1 = (const float*)d_in[49]; const float* b1 = (const float*)d_in[50];
  const float* W2 = (const float*)d_in[51]; const float* b2 = (const float*)d_in[52];
  const float* W3 = (const float*)d_in[53]; const float* b3 = (const float*)d_in[54];
  const float* gG = (const float*)d_in[55]; const float* bE = (const float*)d_in[56];

  gemm(Abufs[0], W1, b1, 16, 8192, 8192, 2048);
  { size_t tot = (size_t)16 * 2048; k_cvt<<<nblk(tot), 256, 0, stream>>>(Outbuf, Abufs[1], tot); }
  gemm(Abufs[1], W2, b2, 16, 2048, 2048, 1024);
  { size_t tot = (size_t)16 * 1024; k_cvt<<<nblk(tot), 256, 0, stream>>>(Outbuf, Abufs[0], tot); }
  gemm(Abufs[0], W3, b3, 16, 1024, 1024, 512);
  k_finalbn<<<2, 256, 0, stream>>>(Outbuf, gG, bE, (float*)d_out);
}
